// SlotAttention_59957743452422
// MI455X (gfx1250) — matrix operand order repaired and checked by reading, not compiled, not run on hardware
//
#include <hip/hip_runtime.h>

// Problem constants (from reference)
#define B_  1024
#define S_  10
#define N_  36
#define F_  2048
#define D_  2176
#define H_  2176

typedef __attribute__((ext_vector_type(16))) __bf16 bf16x16;
typedef __attribute__((ext_vector_type(8)))  float  floatx8;

// ---------------------------------------------------------------------------
// Elementwise helpers
// ---------------------------------------------------------------------------
__global__ void cvt_f32_bf16(const float* __restrict__ src, __bf16* __restrict__ dst, size_t n) {
  size_t i = (size_t)blockIdx.x * blockDim.x + threadIdx.x;
  size_t stride = (size_t)gridDim.x * blockDim.x;
  for (; i < n; i += stride) dst[i] = (__bf16)src[i];
}

__global__ void copy_f32(const float* __restrict__ src, float* __restrict__ dst, size_t n) {
  size_t i = (size_t)blockIdx.x * blockDim.x + threadIdx.x;
  size_t stride = (size_t)gridDim.x * blockDim.x;
  for (; i < n; i += stride) dst[i] = src[i];
}

// ---------------------------------------------------------------------------
// LayerNorm over first nfeat columns + raw copy of ntail trailing columns.
// One block (256 thr = 8 wave32) per row.
// ---------------------------------------------------------------------------
__global__ __launch_bounds__(256)
void ln_kernel(const float* __restrict__ in, int ldin,
               float* __restrict__ out, int ldout,
               const float* __restrict__ g, const float* __restrict__ b,
               int nfeat, int ntail) {
  const int row = blockIdx.x;
  const float* x = in + (size_t)row * ldin;
  float* y = out + (size_t)row * ldout;
  const int tid = threadIdx.x;

  float s = 0.0f, s2 = 0.0f;
  for (int f = tid; f < nfeat; f += 256) { float v = x[f]; s += v; s2 += v * v; }
  for (int off = 16; off; off >>= 1) {
    s  += __shfl_down(s,  off, 32);
    s2 += __shfl_down(s2, off, 32);
  }
  __shared__ float rs[8], rs2[8];
  const int wave = tid >> 5, lane = tid & 31;
  if (lane == 0) { rs[wave] = s; rs2[wave] = s2; }
  __syncthreads();
  if (tid == 0) {
    float t = 0.0f, t2 = 0.0f;
    for (int w = 0; w < 8; ++w) { t += rs[w]; t2 += rs2[w]; }
    rs[0] = t; rs2[0] = t2;
  }
  __syncthreads();
  const float mean = rs[0] / (float)nfeat;
  const float var  = rs2[0] / (float)nfeat - mean * mean;
  const float rstd = rsqrtf(var + 1e-5f);
  for (int f = tid; f < nfeat; f += 256)
    y[f] = (x[f] - mean) * rstd * g[f] + b[f];
  for (int f = tid; f < ntail; f += 256)
    y[nfeat + f] = x[nfeat + f];
}

// ---------------------------------------------------------------------------
// GEMM: C[m,n] = sum_k A[m,k] * W[n,k] + bias[n] (+residual) (opt ReLU)
// A: f32 (row stride lda), W: bf16 (N x K row-major), accum f32 via
// v_wmma_f32_16x16x32_bf16.
// Block tile 128(M) x 128(N), BK=32, 256 threads (8 waves as 4M x 2N).
// Each wave: 32x64 strip = 2 A-frags x 4 B-frags = 8 WMMA / K-step.
// Double-buffered LDS, one barrier per K-step, global prefetch in registers.
// Requires M%128==0, N%128==0, K%32==0 (true for every call site here).
// ---------------------------------------------------------------------------
__global__ __launch_bounds__(256)
void gemm_bf16_wmma(const float* __restrict__ A, int lda,
                    const __bf16* __restrict__ W,
                    const float* __restrict__ bias,
                    float* __restrict__ C, int ldc,
                    const float* __restrict__ residual, int ldr,
                    int M, int N, int K, int do_relu) {
  constexpr int LDSR = 40;  // halfs per row (32 + pad), 16B-aligned rows
  __shared__ __bf16 As[2][128 * LDSR];  // 2 x 10 KB
  __shared__ __bf16 Bs[2][128 * LDSR];  // 2 x 10 KB

  const int tid  = threadIdx.x;
  const int lane = tid & 31;
  const int wave = tid >> 5;
  const int mw   = wave & 3;   // M sub-tile (32 rows)
  const int nw   = wave >> 2;  // N half (64 cols)
  const int row0 = blockIdx.y * 128;
  const int col0 = blockIdx.x * 128;

  // staging assignments: each thread owns 16 contiguous elements of one row
  const int ar = tid >> 1;          // A row 0..127
  const int ac = (tid & 1) << 4;    // A k-chunk 0/16
  const int bn = tid >> 1;          // W row (output col) 0..127
  const int bc = (tid & 1) << 4;    // W k-chunk 0/16

  const float*  aptr = A + (size_t)(row0 + ar) * lda + ac;
  const __bf16* wptr = W + (size_t)(col0 + bn) * K + bc;

  // fragment lane mapping (16-bit A 16x32 / B 32x16, both in contiguous k = kadd + i)
  const int m16  = lane & 15;
  const int koff = (lane >> 4) << 3;   // unused (A also uses kadd)
  const int kadd = (lane >> 4) << 4;   // B: 0 or 16

  floatx8 acc[2][4];
#pragma unroll
  for (int m = 0; m < 2; ++m)
#pragma unroll
    for (int ns = 0; ns < 4; ++ns)
#pragma unroll
      for (int j = 0; j < 8; ++j) acc[m][ns][j] = 0.0f;

  // prologue: stage K-step 0 into buffer 0
  float  areg[16];
  __bf16 breg[16];
#pragma unroll
  for (int i = 0; i < 16; ++i) areg[i] = aptr[i];
#pragma unroll
  for (int i = 0; i < 16; ++i) breg[i] = wptr[i];
  {
    __bf16* ad = &As[0][ar * LDSR + ac];
#pragma unroll
    for (int i = 0; i < 16; ++i) ad[i] = (__bf16)areg[i];
    __bf16* bd = &Bs[0][bn * LDSR + bc];
#pragma unroll
    for (int i = 0; i < 16; ++i) bd[i] = breg[i];
  }
  __syncthreads();

  const int nsteps = K >> 5;
  for (int s = 0; s < nsteps; ++s) {
    const int cur = s & 1;
    const int nxt = cur ^ 1;
    const bool more = (s + 1 < nsteps);

    // prefetch next global tile into registers (overlaps WMMA below)
    if (more) {
      const float*  ap = aptr + (s + 1) * 32;
      const __bf16* wp = wptr + (s + 1) * 32;
#pragma unroll
      for (int i = 0; i < 16; ++i) areg[i] = ap[i];
#pragma unroll
      for (int i = 0; i < 16; ++i) breg[i] = wp[i];
    }

    // A fragments for this wave's 32-row strip
    bf16x16 af[2];
#pragma unroll
    for (int m = 0; m < 2; ++m) {
      const __bf16* arow = &As[cur][(mw * 32 + m * 16 + m16) * LDSR];
#pragma unroll
      for (int i = 0; i < 16; ++i) {
        af[m][i]     = arow[kadd + i];
      }
    }
    // 4 B fragments x 2 A fragments = 8 WMMAs
#pragma unroll
    for (int ns = 0; ns < 4; ++ns) {
      bf16x16 bf;
      const __bf16* brow = &Bs[cur][(nw * 64 + ns * 16 + m16) * LDSR + kadd];
#pragma unroll
      for (int i = 0; i < 16; ++i) bf[i] = brow[i];
      acc[0][ns] = __builtin_amdgcn_wmma_f32_16x16x32_bf16(
          false, af[0], false, bf, (short)0, acc[0][ns], false, false);
      acc[1][ns] = __builtin_amdgcn_wmma_f32_16x16x32_bf16(
          false, af[1], false, bf, (short)0, acc[1][ns], false, false);
    }

    // store prefetched tile to the other buffer
    if (more) {
      __bf16* ad = &As[nxt][ar * LDSR + ac];
#pragma unroll
      for (int i = 0; i < 16; ++i) ad[i] = (__bf16)areg[i];
      __bf16* bd = &Bs[nxt][bn * LDSR + bc];
#pragma unroll
      for (int i = 0; i < 16; ++i) bd[i] = breg[i];
    }
    __syncthreads();
  }

  // epilogue: C/D layout -> row = r + (lane>>4)*8, col = lane&15
  const int rbase = (lane >> 4) << 3;
  const int cidx  = lane & 15;
#pragma unroll
  for (int m = 0; m < 2; ++m) {
#pragma unroll
    for (int ns = 0; ns < 4; ++ns) {
      const int gcol = col0 + nw * 64 + ns * 16 + cidx;
      const float bval = bias ? bias[gcol] : 0.0f;
#pragma unroll
      for (int r = 0; r < 8; ++r) {
        const int grow = row0 + mw * 32 + m * 16 + rbase + r;
        float v = acc[m][ns][r] + bval;
        if (residual) v += residual[(size_t)grow * ldr + gcol];
        if (do_relu) v = fmaxf(v, 0.0f);
        C[(size_t)grow * ldc + gcol] = v;
      }
    }
  }
}

// ---------------------------------------------------------------------------
// Per-batch: dots = q.k^T * scale, mask rows, softmax over SLOT axis (i),
// attn out, updates = attn @ v. One block per batch, 256 threads.
// ---------------------------------------------------------------------------
__global__ __launch_bounds__(256)
void attn_update_kernel(const float* __restrict__ q,     // (B,S,D)
                        const float* __restrict__ kbuf,  // (B,N,D)
                        const float* __restrict__ vbuf,  // (B,N,D)
                        const unsigned char* __restrict__ mask,  // (B,S) bool
                        float* __restrict__ attn_out,    // (B,S,N)
                        float* __restrict__ updates) {   // (B,S,D)
  const int b = blockIdx.x;
  const int tid = threadIdx.x, lane = tid & 31, wave = tid >> 5;
  __shared__ float dots[S_][N_];
  __shared__ float attn[S_][N_];
  const float scale = rsqrtf((float)D_);

  const float* qb = q    + (size_t)b * S_ * D_;
  const float* kb = kbuf + (size_t)b * N_ * D_;

  // dots: one wave per (i,j) pair, lanes reduce over D
  for (int p = wave; p < S_ * N_; p += 8) {
    const int i = p / N_, j = p - i * N_;
    const float* qr = qb + (size_t)i * D_;
    const float* kr = kb + (size_t)j * D_;
    float acc = 0.0f;
    for (int d = lane; d < D_; d += 32) acc += qr[d] * kr[d];
    for (int off = 16; off; off >>= 1) acc += __shfl_down(acc, off, 32);
    if (lane == 0) dots[i][j] = acc * scale;
  }
  __syncthreads();

  // masked softmax over i (slot axis), per column j; write attn output
  if (tid < N_) {
    const int j = tid;
    float col[S_];
    float mx = -INFINITY;
#pragma unroll
    for (int i = 0; i < S_; ++i) {
      float x = mask[(size_t)b * S_ + i] ? -INFINITY : dots[i][j];
      col[i] = x;
      mx = fmaxf(mx, x);
    }
    float sum = 0.0f;
#pragma unroll
    for (int i = 0; i < S_; ++i) { float e = expf(col[i] - mx); col[i] = e; sum += e; }
    const float inv = 1.0f / sum;
#pragma unroll
    for (int i = 0; i < S_; ++i) {
      const float a = col[i] * inv;
      attn[i][j] = a;
      attn_out[((size_t)b * S_ + i) * N_ + j] = a;
    }
  }
  __syncthreads();

  // updates[i,d] = sum_j attn[i][j] * v[j][d]
  const float* vb = vbuf + (size_t)b * N_ * D_;
  for (int d = tid; d < D_; d += 256) {
    float acc[S_];
#pragma unroll
    for (int i = 0; i < S_; ++i) acc[i] = 0.0f;
    for (int j = 0; j < N_; ++j) {
      const float vv = vb[(size_t)j * D_ + d];
#pragma unroll
      for (int i = 0; i < S_; ++i) acc[i] += attn[i][j] * vv;
    }
#pragma unroll
    for (int i = 0; i < S_; ++i)
      updates[((size_t)b * S_ + i) * D_ + d] = acc[i];
  }
}

// ---------------------------------------------------------------------------
// GRU cell elementwise: h_new = (1-z)*n + z*h_prev
// ---------------------------------------------------------------------------
__global__ void gru_kernel(const float* __restrict__ gi,     // (BS,3F)
                           const float* __restrict__ gh,     // (BS,3F)
                           const float* __restrict__ slots,  // (BS,D) h in [:F]
                           float* __restrict__ hnew,         // (BS,F)
                           size_t total) {                   // BS*F
  size_t idx = (size_t)blockIdx.x * blockDim.x + threadIdx.x;
  const size_t stride = (size_t)gridDim.x * blockDim.x;
  for (; idx < total; idx += stride) {
    const size_t row = idx / F_;
    const int f = (int)(idx - row * F_);
    const float* gir = gi + row * (size_t)(3 * F_);
    const float* ghr = gh + row * (size_t)(3 * F_);
    const float r = 1.0f / (1.0f + expf(-(gir[f] + ghr[f])));
    const float z = 1.0f / (1.0f + expf(-(gir[F_ + f] + ghr[F_ + f])));
    const float n = tanhf(gir[2 * F_ + f] + r * ghr[2 * F_ + f]);
    const float hp = slots[row * (size_t)D_ + f];
    hnew[idx] = (1.0f - z) * n + z * hp;
  }
}

// ---------------------------------------------------------------------------
// Host launch
// ---------------------------------------------------------------------------
extern "C" void kernel_launch(void* const* d_in, const int* in_sizes, int n_in,
                              void* d_out, int out_size, void* d_ws, size_t ws_size,
                              hipStream_t stream) {
  (void)in_sizes; (void)n_in; (void)out_size; (void)ws_size;
  const float* cand = (const float*)d_in[0];
  const float* pano = (const float*)d_in[1];
  const unsigned char* mask = (const unsigned char*)d_in[2];  // jnp bool_ = 1 byte
  const float* Wq = (const float*)d_in[3];  const float* bq = (const float*)d_in[4];
  const float* Wk = (const float*)d_in[5];  const float* bk = (const float*)d_in[6];
  const float* Wv = (const float*)d_in[7];  const float* bv = (const float*)d_in[8];
  const float* Wih = (const float*)d_in[9]; const float* bih = (const float*)d_in[10];
  const float* Whh = (const float*)d_in[11];const float* bhh = (const float*)d_in[12];
  const float* W1 = (const float*)d_in[13]; const float* b1 = (const float*)d_in[14];
  const float* W2 = (const float*)d_in[15]; const float* b2 = (const float*)d_in[16];
  const float* ln_in_g = (const float*)d_in[17];   const float* ln_in_b = (const float*)d_in[18];
  const float* ln_sl_g = (const float*)d_in[19];   const float* ln_sl_b = (const float*)d_in[20];
  const float* ln_pr_g = (const float*)d_in[21];   const float* ln_pr_b = (const float*)d_in[22];

  float* out_slots = (float*)d_out;                                   // (B,S,D)
  float* out_attn  = out_slots + (size_t)B_ * S_ * D_;                // (B,S,N)

  // workspace arenas (with reuse; peak ~1.7 GB)
  size_t off = 0;
  auto arena = [&](size_t bytes) {
    void* p = (char*)d_ws + off;
    off += (bytes + 255) & ~(size_t)255;
    return p;
  };
  __bf16* Wqb  = (__bf16*)arena((size_t)D_ * D_ * 2);
  __bf16* Wkb  = (__bf16*)arena((size_t)D_ * D_ * 2);
  __bf16* Wvb  = (__bf16*)arena((size_t)D_ * D_ * 2);
  __bf16* Wihb = (__bf16*)arena((size_t)3 * F_ * D_ * 2);
  __bf16* Whhb = (__bf16*)arena((size_t)3 * F_ * F_ * 2);
  __bf16* W1b  = (__bf16*)arena((size_t)H_ * F_ * 2);
  __bf16* W2b  = (__bf16*)arena((size_t)F_ * H_ * 2);
  float* kbuf  = (float*)arena((size_t)B_ * N_ * D_ * 4);
  float* vbuf  = (float*)arena((size_t)B_ * N_ * D_ * 4);
  float* panoN = (float*)arena((size_t)B_ * N_ * D_ * 4);  // reused per-iter:
  float* sbuf  = panoN;                                    //   s (BS,D)
  float* qbuf  = sbuf + (size_t)B_ * S_ * D_;              //   q (BS,D)
  float* upd   = qbuf + (size_t)B_ * S_ * D_;              //   updates (BS,D)
  float* gi    = (float*)arena((size_t)B_ * S_ * 3 * F_ * 4);
  float* gh    = (float*)arena((size_t)B_ * S_ * 3 * F_ * 4);
  float* mlp1  = gi;                                       // reuse after GRU
  float* pre   = gh;                                       // reuse after GRU
  float* hnew  = (float*)arena((size_t)B_ * S_ * F_ * 4);

  const int BS = B_ * S_;
  const int BN = B_ * N_;

  // 1) weights -> bf16
  auto cvt = [&](const float* s, __bf16* d, size_t n) {
    cvt_f32_bf16<<<dim3(4096), dim3(256), 0, stream>>>(s, d, n);
  };
  cvt(Wq,  Wqb,  (size_t)D_ * D_);
  cvt(Wk,  Wkb,  (size_t)D_ * D_);
  cvt(Wv,  Wvb,  (size_t)D_ * D_);
  cvt(Wih, Wihb, (size_t)3 * F_ * D_);
  cvt(Whh, Whhb, (size_t)3 * F_ * F_);
  cvt(W1,  W1b,  (size_t)H_ * F_);
  cvt(W2,  W2b,  (size_t)F_ * H_);

  // 2) slots <- cand_feat (angle columns persist in d_out for all iterations)
  copy_f32<<<dim3(8192), dim3(256), 0, stream>>>(cand, out_slots, (size_t)BS * D_);

  // 3) pano = concat(LN(pano[:, :F]), pano[:, F:])
  ln_kernel<<<dim3(BN), dim3(256), 0, stream>>>(pano, D_, panoN, D_,
                                                ln_in_g, ln_in_b, F_, D_ - F_);

  // 4) k, v GEMMs  (M=36864, N=2176, K=2176)
  gemm_bf16_wmma<<<dim3(D_ / 128, BN / 128), dim3(256), 0, stream>>>(
      panoN, D_, Wkb, bk, kbuf, D_, nullptr, 0, BN, D_, D_, 0);
  gemm_bf16_wmma<<<dim3(D_ / 128, BN / 128), dim3(256), 0, stream>>>(
      panoN, D_, Wvb, bv, vbuf, D_, nullptr, 0, BN, D_, D_, 0);

  // 5) three slot-attention iterations
  for (int it = 0; it < 3; ++it) {
    // s = concat(LN(h_prev), angle)
    ln_kernel<<<dim3(BS), dim3(256), 0, stream>>>(out_slots, D_, sbuf, D_,
                                                  ln_sl_g, ln_sl_b, F_, D_ - F_);
    // q = s @ Wq^T + bq
    gemm_bf16_wmma<<<dim3(D_ / 128, BS / 128), dim3(256), 0, stream>>>(
        sbuf, D_, Wqb, bq, qbuf, D_, nullptr, 0, BS, D_, D_, 0);
    // dots/softmax(slot axis)/updates; writes attn to d_out (last iter wins)
    attn_update_kernel<<<dim3(B_), dim3(256), 0, stream>>>(
        qbuf, kbuf, vbuf, mask, out_attn, upd);
    // gi = updates @ W_ih^T + b_ih   (K=2176 -> N=6144)
    gemm_bf16_wmma<<<dim3(3 * F_ / 128, BS / 128), dim3(256), 0, stream>>>(
        upd, D_, Wihb, bih, gi, 3 * F_, nullptr, 0, BS, 3 * F_, D_, 0);
    // gh = h_prev @ W_hh^T + b_hh    (A = slots[:, :F], strided lda=D)
    gemm_bf16_wmma<<<dim3(3 * F_ / 128, BS / 128), dim3(256), 0, stream>>>(
        out_slots, D_, Whhb, bhh, gh, 3 * F_, nullptr, 0, BS, 3 * F_, F_, 0);
    // GRU gates -> hnew
    gru_kernel<<<dim3(20480), dim3(256), 0, stream>>>(
        gi, gh, out_slots, hnew, (size_t)BS * F_);
    // pre = LN(hnew)
    ln_kernel<<<dim3(BS), dim3(256), 0, stream>>>(hnew, F_, pre, F_,
                                                  ln_pr_g, ln_pr_b, F_, 0);
    // mlp1 = relu(pre @ W1^T + b1)
    gemm_bf16_wmma<<<dim3(H_ / 128, BS / 128), dim3(256), 0, stream>>>(
        pre, F_, W1b, b1, mlp1, H_, nullptr, 0, BS, H_, F_, 1);
    // slots[:, :F] = hnew + mlp1 @ W2^T + b2  (residual + strided store)
    gemm_bf16_wmma<<<dim3(F_ / 128, BS / 128), dim3(256), 0, stream>>>(
        mlp1, H_, W2b, b2, out_slots, D_, hnew, F_, BS, F_, H_, 0);
  }
}
